// PerformerAttention_15771119911066
// MI455X (gfx1250) — matrix operand order repaired and checked by reading, not compiled, not run on hardware
//
#include <hip/hip_runtime.h>
#include <hip/hip_bf16.h>
#include <cstdint>

typedef __attribute__((ext_vector_type(16))) _Float16 v16h;
typedef __attribute__((ext_vector_type(8)))  float    v8f;

// ---------------- constants ----------------
// B=8, N=4096, C=768, H=12, d=64, M=256, R=B*N=32768, 3C=2304
#define R_ROWS   32768
#define CC       768
#define C3       2304
#define HEADS    12
#define DH       64
#define MM       256
#define NB       8
#define NSEQ     4096
#define BH       (NB*HEADS)            // 96
#define NORM     0.35355339059327373f  // 64^-0.25
#define RATIO    0.0625f               // 256^-0.5
#define EPSF     1e-4f

// ---------------- workspace layout (bytes) ----------------
static constexpr size_t OFF_XH     = 0;                     // R*C f16      = 50,331,648
static constexpr size_t OFF_WQKVH  = 50331648ull;           // 2304*768 f16 =  3,538,944
static constexpr size_t OFF_WPROJH = 53870592ull;           // 768*768 f16  =  1,179,648
static constexpr size_t OFF_PROJH  = 55050240ull;           // 256*64 f16   =     32,768
static constexpr size_t OFF_QKVH   = 55083008ull;           // R*2304 f16   =150,994,944
static constexpr size_t OFF_DIAGQ  = 206077952ull;          // B*H*N f32    =  1,572,864
static constexpr size_t OFF_DIAGK  = 207650816ull;          // B*H*N f32    =  1,572,864
static constexpr size_t OFF_KMAX   = 209223680ull;          // 8 u32 (pad 256)
static constexpr size_t OFF_KSUM   = 209223936ull;          // B*H*M f32    =     98,304
static constexpr size_t OFF_CTX    = 209322240ull;          // B*H*M*d f32  =  6,291,456
static constexpr size_t OFF_CTXTH  = 215613696ull;          // B*H*d*M f16  =  3,145,728
static constexpr size_t OFF_ATTNH  = 218759424ull;          // R*C f16      = 50,331,648

// ---------------- helpers ----------------
__device__ __forceinline__ v8f wmma_f16(v16h a, v16h b, v8f c) {
  return __builtin_amdgcn_wmma_f32_16x16x32_f16(false, a, false, b, (short)0, c, false, false);
}

// A-fragment (16x32 f16): lane<16 -> K 0..7 & 16..23 ; lane>=16 -> K 8..15 & 24..31
__device__ __forceinline__ v16h ldfragA(const _Float16* p, int lane) {
  const _Float16* p0 = p + ((lane & 16) ? 8 : 0);
  union { float4 f[2]; v16h v; } u;
  u.f[0] = *reinterpret_cast<const float4*>(p0);
  u.f[1] = *reinterpret_cast<const float4*>(p0 + 16);
  return u.v;
}

// B-fragment (32x16 f16), column-contiguous source (row of W): lane<16 -> K 0..7 & 16..23 ; lane>=16 -> K 8..15 & 24..31
__device__ __forceinline__ v16h ldfragB(const _Float16* p, int lane) {
  const _Float16* p0 = p + ((lane & 16) ? 8 : 0);
  union { float4 f[2]; v16h v; } u;
  u.f[0] = *reinterpret_cast<const float4*>(p0);
  u.f[1] = *reinterpret_cast<const float4*>(p0 + 16);
  return u.v;
}

// async copy 16B from global to LDS (tracked on ASYNCcnt)
__device__ __forceinline__ void async_cp16(unsigned lds_off_bytes, const void* gaddr) {
  asm volatile("global_load_async_to_lds_b128 %0, %1, off"
               :: "v"(lds_off_bytes), "v"(gaddr) : "memory");
}
__device__ __forceinline__ void wait_async0() {
  asm volatile("s_wait_asynccnt 0x0" ::: "memory");
}

__device__ __forceinline__ unsigned enc_f(float f) {
  unsigned u = __float_as_uint(f);
  return (u & 0x80000000u) ? ~u : (u | 0x80000000u);
}
__device__ __forceinline__ float dec_f(unsigned u) {
  return (u & 0x80000000u) ? __uint_as_float(u ^ 0x80000000u) : __uint_as_float(~u);
}

// ---------------- kernel 0: convert + init ----------------
__global__ void k_setup(const float* __restrict__ x, const float* __restrict__ wqkv,
                        const float* __restrict__ wproj, const float* __restrict__ proj,
                        _Float16* xh, _Float16* wqkvh, _Float16* wprojh, _Float16* projh,
                        float* ksum, float* ctx, unsigned* kmax_enc) {
  size_t i0 = (size_t)blockIdx.x * blockDim.x + threadIdx.x;
  size_t st = (size_t)gridDim.x * blockDim.x;
  for (size_t i = i0; i < 25165824ull; i += st) xh[i]    = (_Float16)x[i];
  for (size_t i = i0; i < 1769472ull;  i += st) wqkvh[i] = (_Float16)wqkv[i];
  for (size_t i = i0; i < 589824ull;   i += st) wprojh[i]= (_Float16)wproj[i];
  for (size_t i = i0; i < 16384ull;    i += st) projh[i] = (_Float16)(proj[i] * NORM);
  for (size_t i = i0; i < 24576ull;    i += st) ksum[i]  = 0.f;
  for (size_t i = i0; i < 1572864ull;  i += st) ctx[i]   = 0.f;
  for (size_t i = i0; i < 8ull;        i += st) kmax_enc[i] = 0u;
}

// ---------------- kernel 1: qkv = x @ wqkv^T (f16 out), async-LDS staged B ----------------
#define BS_PITCH 40   // halves per column slice (32 used, 80B pitch -> conflict-free frag reads)
__global__ __launch_bounds__(256) void k_gemm_qkv(const _Float16* __restrict__ xh,
                                                  const _Float16* __restrict__ wh,
                                                  _Float16* __restrict__ qkvh) {
  __shared__ _Float16 bs[2][128 * BS_PITCH];   // 2 x 10KB
  const int lane = threadIdx.x & 31, wave = threadIdx.x >> 5, lr = lane & 15;
  const int half8 = (lane & 16) ? 8 : 0;
  const int rowblk = blockIdx.x / 18, colblk = blockIdx.x % 18;
  const int rowbase = rowblk * 128 + wave * 16;
  const int colbase = colblk * 128;

  // stage B k-slice: 128 cols x 32 halves (64B per col) = 512 x 16B chunks
  auto stage = [&](int kk, int buf) {
#pragma unroll
    for (int i = 0; i < 2; i++) {
      int f = (int)threadIdx.x + i * 256;
      int col = f >> 2, ch = f & 3;
      const _Float16* g = wh + (size_t)(colbase + col) * CC + kk + ch * 8;
      unsigned lds = (unsigned)(uintptr_t)(&bs[buf][col * BS_PITCH + ch * 8]);
      async_cp16(lds, g);
    }
  };

  v8f acc[8] = {};
  const _Float16* arow = xh + (size_t)(rowbase + lr) * CC;
  stage(0, 0);
  int buf = 0;
  for (int kk = 0; kk < CC; kk += 32) {
    wait_async0();
    __syncthreads();
    if (kk + 32 < CC) stage(kk + 32, buf ^ 1);
    v16h A = ldfragA(arow + kk, lane);
#pragma unroll
    for (int t = 0; t < 8; t++) {
      v16h Bf = ldfragB(&bs[buf][(t * 16 + lr) * BS_PITCH], lane);
      acc[t] = wmma_f16(A, Bf, acc[t]);
    }
    buf ^= 1;
  }
#pragma unroll
  for (int t = 0; t < 8; t++) {
    int col = colbase + t * 16 + lr;
#pragma unroll
    for (int r = 0; r < 8; r++) {
      int row = rowbase + r + half8;
      qkvh[(size_t)row * C3 + col] = (_Float16)acc[t][r];
    }
  }
}

// ---------------- kernel 2: diag sums ----------------
__global__ void k_diag(const _Float16* __restrict__ qkvh, float* diagq, float* diagk) {
  int idx = blockIdx.x * blockDim.x + threadIdx.x;
  if (idx >= 786432) return;
  int mat = idx / 393216;
  int r = idx - mat * 393216;
  int b = r / 49152; int rem = r % 49152; int h = rem >> 12; int n = rem & 4095;
  const _Float16* p = qkvh + (size_t)(b * NSEQ + n) * C3 + mat * CC + h * DH;
  float s = 0.f;
  for (int i = 0; i < DH; i++) { float v = (float)p[i]; s += v * v; }
  float* dst = mat ? diagk : diagq;
  dst[(size_t)(b * HEADS + h) * NSEQ + n] = 0.0625f * s;  // 0.5 * d^-0.5
}

// ---------------- kernel 3: per-batch max of k_dash ----------------
__global__ __launch_bounds__(256) void k_kmax(const _Float16* __restrict__ qkvh,
                                              const _Float16* __restrict__ projh,
                                              unsigned* kmax_enc) {
  const int lane = threadIdx.x & 31, wave = threadIdx.x >> 5, lr = lane & 15;
  int bh = blockIdx.x >> 5, nblk = blockIdx.x & 31;
  int b = bh / HEADS, h = bh % HEADS;
  int n0 = nblk * 128 + wave * 16;
  v8f acc[16] = {};
  const _Float16* arow = qkvh + (size_t)(b * NSEQ + n0 + lr) * C3 + CC + h * DH;
  for (int kk = 0; kk < DH; kk += 32) {
    v16h A = ldfragA(arow + kk, lane);
#pragma unroll
    for (int t = 0; t < 16; t++) {
      v16h Bf = ldfragB(projh + (t * 16 + lr) * DH + kk, lane);
      acc[t] = wmma_f16(A, Bf, acc[t]);
    }
  }
  float m = -3.4e38f;
#pragma unroll
  for (int t = 0; t < 16; t++)
#pragma unroll
    for (int r = 0; r < 8; r++) m = fmaxf(m, acc[t][r]);
  for (int s = 1; s < 32; s <<= 1) m = fmaxf(m, __shfl_xor(m, s));
  __shared__ float smax[8];
  if (lane == 0) smax[wave] = m;
  __syncthreads();
  if (threadIdx.x == 0) {
    float mm = smax[0];
    for (int i = 1; i < 8; i++) mm = fmaxf(mm, smax[i]);
    atomicMax(&kmax_enc[b], enc_f(mm));
  }
}

// ---------------- kernel 4: k' -> ksum partial + context partial ----------------
#define KP_PITCH 264
#define VS_PITCH 72
__global__ __launch_bounds__(256) void k_kprime_ctx(const _Float16* __restrict__ qkvh,
                                                    const _Float16* __restrict__ projh,
                                                    const float* __restrict__ diagk,
                                                    const unsigned* __restrict__ kmax_enc,
                                                    float* ksum, float* ctx) {
  __shared__ _Float16 kp[64 * KP_PITCH];
  __shared__ _Float16 vs[64 * VS_PITCH];
  __shared__ float ksum_s[MM];
  const int lane = threadIdx.x & 31, wave = threadIdx.x >> 5, lr = lane & 15;
  const int half8 = (lane & 16) ? 8 : 0;
  int bh = blockIdx.x >> 4, cblk = blockIdx.x & 15;   // 16 blocks per (b,h), 256 rows each
  int b = bh / HEADS, h = bh % HEADS;
  float mval = dec_f(kmax_enc[b]);
  ksum_s[threadIdx.x] = 0.f;
  v8f acc2[2][4] = {};
  for (int ch = 0; ch < 4; ch++) {
    int nbase = cblk * 256 + ch * 64;
    __syncthreads();
    if (wave < 4) {
      int n0 = nbase + wave * 16;
      v8f acc[16] = {};
      const _Float16* arow = qkvh + (size_t)(b * NSEQ + n0 + lr) * C3 + CC + h * DH;
      for (int kk = 0; kk < DH; kk += 32) {
        v16h A = ldfragA(arow + kk, lane);
#pragma unroll
        for (int t = 0; t < 16; t++) {
          v16h Bf = ldfragB(projh + (t * 16 + lr) * DH + kk, lane);
          acc[t] = wmma_f16(A, Bf, acc[t]);
        }
      }
      float dg[8];
#pragma unroll
      for (int r = 0; r < 8; r++) dg[r] = diagk[(size_t)bh * NSEQ + n0 + r + half8];
#pragma unroll
      for (int t = 0; t < 16; t++) {
        float cs = 0.f;
#pragma unroll
        for (int r = 0; r < 8; r++) {
          float val = RATIO * (__expf(acc[t][r] - dg[r] - mval) + EPSF);
          kp[(wave * 16 + r + half8) * KP_PITCH + t * 16 + lr] = (_Float16)val;
          cs += val;
        }
        cs += __shfl_xor(cs, 16);
        if (lane < 16) atomicAdd(&ksum_s[t * 16 + lane], cs);
      }
    } else {
      // waves 4..7: async DMA of the v chunk (64 rows x 64 halves) into LDS
      int base_f = (wave - 4) * 32 + lane;          // 0..127
#pragma unroll
      for (int i = 0; i < 4; i++) {
        int f = base_f + i * 128;                   // 0..511 (16B chunks)
        int row = f >> 3, chk = f & 7;
        const _Float16* g = qkvh + (size_t)(b * NSEQ + nbase + row) * C3 + 2 * CC + h * DH + chk * 8;
        unsigned lds = (unsigned)(uintptr_t)(&vs[row * VS_PITCH + chk * 8]);
        async_cp16(lds, g);
      }
    }
    wait_async0();
    __syncthreads();
    // context partial: ctx += k'^T @ v ; wave owns M-tiles [2w,2w+1] x d-tiles 0..3
    const int koffB = (lane & 16) ? 8 : 0;
    for (int kc = 0; kc < 64; kc += 32) {
      v16h a[2];
#pragma unroll
      for (int i = 0; i < 2; i++) {
        int m = wave * 32 + i * 16 + lr;
#pragma unroll
        for (int j = 0; j < 8; j++) {
          a[i][j]     = kp[(kc + half8 + j) * KP_PITCH + m];
          a[i][8 + j] = kp[(kc + 16 + half8 + j) * KP_PITCH + m];
        }
      }
#pragma unroll
      for (int dt = 0; dt < 4; dt++) {
        v16h Bf;
        int dd = dt * 16 + lr;
#pragma unroll
        for (int j = 0; j < 16; j++) Bf[j] = vs[(kc + koffB + j + (j & 8)) * VS_PITCH + dd];
#pragma unroll
        for (int i = 0; i < 2; i++) acc2[i][dt] = wmma_f16(a[i], Bf, acc2[i][dt]);
      }
    }
  }
  __syncthreads();
  atomicAdd(&ksum[(size_t)bh * MM + threadIdx.x], ksum_s[threadIdx.x]);
#pragma unroll
  for (int i = 0; i < 2; i++)
#pragma unroll
    for (int dt = 0; dt < 4; dt++)
#pragma unroll
      for (int r = 0; r < 8; r++) {
        int m = wave * 32 + i * 16 + r + half8;
        int dd = dt * 16 + lr;
        atomicAdd(&ctx[((size_t)bh * MM + m) * DH + dd], acc2[i][dt][r]);
      }
}

// ---------------- kernel 5: transpose+convert context ----------------
__global__ void k_ctxcvt(const float* __restrict__ ctx, _Float16* __restrict__ ctxTh) {
  int idx = blockIdx.x * blockDim.x + threadIdx.x;
  if (idx >= 1572864) return;
  int bh = idx / 16384; int rem = idx & 16383; int dd = rem >> 8; int m = rem & 255;
  ctxTh[((size_t)bh * DH + dd) * MM + m] = (_Float16)ctx[((size_t)bh * MM + m) * DH + dd];
}

// ---------------- kernel 6: q' -> dinv -> out = (q' @ ctx) * dinv (f16, head-interleaved) ----------------
#define QP_PITCH 264
__global__ __launch_bounds__(128) void k_qprime(const _Float16* __restrict__ qkvh,
                                                const _Float16* __restrict__ projh,
                                                const float* __restrict__ diagq,
                                                const float* __restrict__ ksum,
                                                const _Float16* __restrict__ ctxTh,
                                                _Float16* __restrict__ attnh) {
  __shared__ _Float16 qp[4 * 16 * QP_PITCH];
  const int lane = threadIdx.x & 31, wave = threadIdx.x >> 5, lr = lane & 15;
  const int half8 = (lane & 16) ? 8 : 0;
  int bh = blockIdx.x >> 6, nblk = blockIdx.x & 63;
  int b = bh / HEADS, h = bh % HEADS;
  int n0 = nblk * 64 + wave * 16;
  // phase A: q_dash
  v8f acc[16] = {};
  const _Float16* arow = qkvh + (size_t)(b * NSEQ + n0 + lr) * C3 + h * DH;
  for (int kk = 0; kk < DH; kk += 32) {
    v16h A = ldfragA(arow + kk, lane);
#pragma unroll
    for (int t = 0; t < 16; t++) {
      v16h Bf = ldfragB(projh + (t * 16 + lr) * DH + kk, lane);
      acc[t] = wmma_f16(A, Bf, acc[t]);
    }
  }
  float dg[8], rm[8], srow[8], dinv[8];
#pragma unroll
  for (int r = 0; r < 8; r++) {
    dg[r] = diagq[(size_t)bh * NSEQ + n0 + r + half8];
    float m = acc[0][r];
#pragma unroll
    for (int t = 1; t < 16; t++) m = fmaxf(m, acc[t][r]);
    for (int s = 1; s < 16; s <<= 1) m = fmaxf(m, __shfl_xor(m, s));  // within lane-half
    rm[r] = m;
    srow[r] = 0.f;
  }
#pragma unroll
  for (int t = 0; t < 16; t++) {
    float ks = ksum[(size_t)bh * MM + t * 16 + lr];
#pragma unroll
    for (int r = 0; r < 8; r++) {
      float val = RATIO * (__expf(acc[t][r] - dg[r] - rm[r]) + EPSF);
      qp[(wave * 16 + r + half8) * QP_PITCH + t * 16 + lr] = (_Float16)val;
      srow[r] += val * ks;
    }
  }
#pragma unroll
  for (int r = 0; r < 8; r++) {
    for (int s = 1; s < 16; s <<= 1) srow[r] += __shfl_xor(srow[r], s);
    dinv[r] = 1.0f / srow[r];
  }
  // phase B: out = q' @ ctx (A from LDS, B from ctxTh rows)
  v8f out4[4] = {};
  for (int kc = 0; kc < MM; kc += 32) {
    v16h A2;
    const _Float16* qrow = qp + (wave * 16 + lr) * QP_PITCH + kc;
#pragma unroll
    for (int j = 0; j < 8; j++) {
      A2[j]     = qrow[half8 + j];
      A2[8 + j] = qrow[16 + half8 + j];
    }
#pragma unroll
    for (int dt = 0; dt < 4; dt++) {
      v16h Bf = ldfragB(ctxTh + ((size_t)bh * DH + dt * 16 + lr) * MM + kc, lane);
      out4[dt] = wmma_f16(A2, Bf, out4[dt]);
    }
  }
#pragma unroll
  for (int dt = 0; dt < 4; dt++)
#pragma unroll
    for (int r = 0; r < 8; r++) {
      int n = n0 + r + half8;
      attnh[(size_t)(b * NSEQ + n) * CC + h * DH + dt * 16 + lr] = (_Float16)(out4[dt][r] * dinv[r]);
    }
}

// ---------------- kernel 7: out = attn @ wproj^T + b (f32 out), async-LDS staged B ----------------
__global__ __launch_bounds__(256) void k_final(const _Float16* __restrict__ attnh,
                                               const _Float16* __restrict__ wprojh,
                                               const float* __restrict__ bproj,
                                               float* __restrict__ out) {
  __shared__ _Float16 bs[2][128 * BS_PITCH];
  const int lane = threadIdx.x & 31, wave = threadIdx.x >> 5, lr = lane & 15;
  const int half8 = (lane & 16) ? 8 : 0;
  const int rowblk = blockIdx.x / 6, colblk = blockIdx.x % 6;
  const int rowbase = rowblk * 128 + wave * 16;
  const int colbase = colblk * 128;

  auto stage = [&](int kk, int buf) {
#pragma unroll
    for (int i = 0; i < 2; i++) {
      int f = (int)threadIdx.x + i * 256;
      int col = f >> 2, ch = f & 3;
      const _Float16* g = wprojh + (size_t)(colbase + col) * CC + kk + ch * 8;
      unsigned lds = (unsigned)(uintptr_t)(&bs[buf][col * BS_PITCH + ch * 8]);
      async_cp16(lds, g);
    }
  };

  v8f acc[8] = {};
  const _Float16* arow = attnh + (size_t)(rowbase + lr) * CC;
  stage(0, 0);
  int buf = 0;
  for (int kk = 0; kk < CC; kk += 32) {
    wait_async0();
    __syncthreads();
    if (kk + 32 < CC) stage(kk + 32, buf ^ 1);
    v16h A = ldfragA(arow + kk, lane);
#pragma unroll
    for (int t = 0; t < 8; t++) {
      v16h Bf = ldfragB(&bs[buf][(t * 16 + lr) * BS_PITCH], lane);
      acc[t] = wmma_f16(A, Bf, acc[t]);
    }
    buf ^= 1;
  }
#pragma unroll
  for (int t = 0; t < 8; t++) {
    int col = colbase + t * 16 + lr;
    float bias = bproj[col];
#pragma unroll
    for (int r = 0; r < 8; r++) {
      int row = rowbase + r + half8;
      out[(size_t)row * CC + col] = acc[t][r] + bias;
    }
  }
}

// ---------------- host launcher ----------------
extern "C" void kernel_launch(void* const* d_in, const int* in_sizes, int n_in,
                              void* d_out, int out_size, void* d_ws, size_t ws_size,
                              hipStream_t stream) {
  const float* x     = (const float*)d_in[0];
  const float* wqkv  = (const float*)d_in[1];
  const float* wproj = (const float*)d_in[2];
  const float* bproj = (const float*)d_in[3];
  const float* proj  = (const float*)d_in[4];
  float* out = (float*)d_out;
  char* ws = (char*)d_ws;

  _Float16* xh     = (_Float16*)(ws + OFF_XH);
  _Float16* wqkvh  = (_Float16*)(ws + OFF_WQKVH);
  _Float16* wprojh = (_Float16*)(ws + OFF_WPROJH);
  _Float16* projh  = (_Float16*)(ws + OFF_PROJH);
  _Float16* qkvh   = (_Float16*)(ws + OFF_QKVH);
  float*    diagq  = (float*)   (ws + OFF_DIAGQ);
  float*    diagk  = (float*)   (ws + OFF_DIAGK);
  unsigned* kmaxe  = (unsigned*)(ws + OFF_KMAX);
  float*    ksum   = (float*)   (ws + OFF_KSUM);
  float*    ctx    = (float*)   (ws + OFF_CTX);
  _Float16* ctxTh  = (_Float16*)(ws + OFF_CTXTH);
  _Float16* attnh  = (_Float16*)(ws + OFF_ATTNH);

  k_setup<<<2048, 256, 0, stream>>>(x, wqkv, wproj, proj, xh, wqkvh, wprojh, projh,
                                    ksum, ctx, kmaxe);
  k_gemm_qkv<<<256 * 18, 256, 0, stream>>>(xh, wqkvh, qkvh);
  k_diag<<<3072, 256, 0, stream>>>(qkvh, diagq, diagk);
  k_kmax<<<BH * 32, 256, 0, stream>>>(qkvh, projh, kmaxe);
  k_kprime_ctx<<<BH * 16, 256, 0, stream>>>(qkvh, projh, diagk, kmaxe, ksum, ctx);
  k_ctxcvt<<<6144, 256, 0, stream>>>(ctx, ctxTh);
  k_qprime<<<BH * 64, 128, 0, stream>>>(qkvh, projh, diagq, ksum, ctxTh, attnh);
  k_final<<<256 * 6, 256, 0, stream>>>(attnh, wprojh, bproj, out);
}
